// ToeplitzAttention_28750511079448
// MI455X (gfx1250) — matrix operand order repaired and checked by reading, not compiled, not run on hardware
//
#include <hip/hip_runtime.h>

// ToeplitzAttention fused kernel for gfx1250 (MI455X), wave32 WMMA f16->f32.
// B=8, L=1024, H=8, E=128 (from reference setup_inputs).
// v4: v2 structure (best instruction mix), plus: folded sigmoid constant,
//     pair-packed (half2 / ds_store_b32) V-tile staging, per-element diag
//     cndmask (no divergent branch duplication). Latency hiding comes from
//     multi-wave occupancy, not intra-wave pipelining (VGPR-bound at ~240).

typedef __attribute__((ext_vector_type(16))) _Float16 v16h;
typedef __attribute__((ext_vector_type(8)))  _Float16 v8h;
typedef __attribute__((ext_vector_type(8)))  float    v8f;

#define BQ 8
#define LQ 1024
#define NH 8
#define ED 128
#define VSTRIDE 40    // V tile [e][s] padded row stride (halves): 80B, 16B aligned
#define KSTRIDE 136   // K/Q tile [s][e] padded row stride (halves): 272B -> 4-bank row skew

union H16 { v16h v; _Float16 h[16]; };
union H8  { v8h  v; _Float16 h[8];  };
union HP  { _Float16 h[2]; unsigned int u; };

__device__ __forceinline__ void cvt8(_Float16* d, float4 a, float4 b) {
  d[0] = (_Float16)a.x; d[1] = (_Float16)a.y; d[2] = (_Float16)a.z; d[3] = (_Float16)a.w;
  d[4] = (_Float16)b.x; d[5] = (_Float16)b.y; d[6] = (_Float16)b.z; d[7] = (_Float16)b.w;
}

// Two 16B LDS loads -> one 16x16x32 A/B operand (halves [0..7] and [8..15]).
__device__ __forceinline__ v16h ldsOp(const _Float16* base, int second_off_halves) {
  H16 t;
  *reinterpret_cast<v8h*>(&t.h[0]) = *reinterpret_cast<const v8h*>(base);
  *reinterpret_cast<v8h*>(&t.h[8]) = *reinterpret_cast<const v8h*>(base + second_off_halves);
  return t.v;
}

__global__ __launch_bounds__(128) void toeplitz_attn_kernel(
    const float* __restrict__ Q, const float* __restrict__ K,
    const float* __restrict__ V, const float* __restrict__ LT,
    float* __restrict__ Vout, float* __restrict__ Aout, float* __restrict__ Ent) {
  __shared__ _Float16 klds[32 * KSTRIDE];    // K s-tile, row-major [s][e], f16
  __shared__ _Float16 qlds[32 * KSTRIDE];    // Q s-tile, row-major [s][e], f16
  __shared__ _Float16 vlds[ED * VSTRIDE];    // V s-tile, transposed [e][s], f16
  __shared__ _Float16 attlds[4][16 * 32];    // per-wave att tile for C->A relayout

  const int tid  = threadIdx.x;
  const int wave = tid >> 5;
  const int lane = tid & 31;
  const int n    = lane & 15;
  const int hi   = lane >> 4;

  const int bt     = blockIdx.x;          // 0..1023
  const int l_band = (bt & 15) * 64;      // 16 bands of 64 rows per (b,h)
  const int h      = (bt >> 4) & 7;
  const int b      = bt >> 7;
  const int l0     = l_band + wave * 16;  // this wave's 16-row band

  float tau = __expf(LT[0]);
  tau = fminf(fmaxf(tau, 0.1f), 5.0f);
  const float nit05 = -0.5f / tau;        // folds the 0.5 split and 1/tau scales

  // ---- Preload Q-row / K-row A-operands (16x128 f16, ISA 16-bit A layout) ----
  // A 16x32: lane m holds row m; halves 0-7 = k in [hi*8, hi*8+8),
  // halves 8-15 = k in [16+hi*8, 16+hi*8+8).
  H16 QA[4], KA[4];
  {
    const float* qrow = Q + (((size_t)b * LQ + (l0 + n)) * NH + h) * ED;
    const float* krow = K + (((size_t)b * LQ + (l0 + n)) * NH + h) * ED;
#pragma unroll
    for (int c = 0; c < 4; ++c) {
      const int e0 = c * 32;
      const float4* q4a = reinterpret_cast<const float4*>(qrow + e0 + hi * 8);
      const float4* q4b = reinterpret_cast<const float4*>(qrow + e0 + 16 + hi * 8);
      cvt8(QA[c].h,     q4a[0], q4a[1]);
      cvt8(QA[c].h + 8, q4b[0], q4b[1]);
      const float4* k4a = reinterpret_cast<const float4*>(krow + e0 + hi * 8);
      const float4* k4b = reinterpret_cast<const float4*>(krow + e0 + 16 + hi * 8);
      cvt8(KA[c].h,     k4a[0], k4a[1]);
      cvt8(KA[c].h + 8, k4b[0], k4b[1]);
    }
  }

  v8f accV[8];
  const v8f vzero = {};
#pragma unroll
  for (int t = 0; t < 8; ++t) accV[t] = vzero;
  float ent[8] = {0.f, 0.f, 0.f, 0.f, 0.f, 0.f, 0.f, 0.f};

  // K/Q staging mapping: thread owns tile-row kr, 32-float e segment
  const int kr = tid >> 2;          // 0..31
  const int ke = (tid & 3) * 32;    // e segment start
  // V staging mapping: thread owns two adjacent tile-rows (half2 packing), 16-float e seg
  const int rp = tid >> 3;          // 0..15 -> rows 2rp, 2rp+1
  const int pe = (tid & 7) * 16;    // e segment start

  float* const aout_base = Aout + (size_t)(b * NH + h) * LQ * LQ;

  for (int s0 = 0; s0 < LQ; s0 += 32) {
    __syncthreads();  // previous step's LDS reads complete

    // ---- Stage K/Q tiles row-major f16 ----
    {
      const size_t rbase = (((size_t)b * LQ + s0 + kr) * NH + h) * ED + ke;
      const float* krow = K + rbase;
      const float* qrow = Q + rbase;
      if (s0 + 32 < LQ) {
        __builtin_prefetch(krow + 32 * NH * ED, 0, 1);
        __builtin_prefetch(qrow + 32 * NH * ED, 0, 1);
      }
#pragma unroll
      for (int i = 0; i < 4; ++i) {
        H8 ck, cq;
        cvt8(ck.h, reinterpret_cast<const float4*>(krow)[2 * i],
                   reinterpret_cast<const float4*>(krow)[2 * i + 1]);
        *reinterpret_cast<v8h*>(&klds[kr * KSTRIDE + ke + 8 * i]) = ck.v;
        cvt8(cq.h, reinterpret_cast<const float4*>(qrow)[2 * i],
                   reinterpret_cast<const float4*>(qrow)[2 * i + 1]);
        *reinterpret_cast<v8h*>(&qlds[kr * KSTRIDE + ke + 8 * i]) = cq.v;
      }
    }
    // ---- Stage V tile transposed f16, two s-rows packed per ds_store_b32 ----
    {
      const float* vrow0 = V + (((size_t)b * LQ + s0 + 2 * rp) * NH + h) * ED + pe;
      const float* vrow1 = vrow0 + NH * ED;  // next s row
      if (s0 + 32 < LQ) __builtin_prefetch(vrow0 + 32 * NH * ED, 0, 1);
#pragma unroll
      for (int i = 0; i < 4; ++i) {
        const float4 f0 = reinterpret_cast<const float4*>(vrow0)[i];
        const float4 f1 = reinterpret_cast<const float4*>(vrow1)[i];
        HP p0, p1, p2, p3;
        p0.h[0] = (_Float16)f0.x; p0.h[1] = (_Float16)f1.x;
        p1.h[0] = (_Float16)f0.y; p1.h[1] = (_Float16)f1.y;
        p2.h[0] = (_Float16)f0.z; p2.h[1] = (_Float16)f1.z;
        p3.h[0] = (_Float16)f0.w; p3.h[1] = (_Float16)f1.w;
        const int e = pe + 4 * i;
        *reinterpret_cast<unsigned int*>(&vlds[(e + 0) * VSTRIDE + 2 * rp]) = p0.u;
        *reinterpret_cast<unsigned int*>(&vlds[(e + 1) * VSTRIDE + 2 * rp]) = p1.u;
        *reinterpret_cast<unsigned int*>(&vlds[(e + 2) * VSTRIDE + 2 * rp]) = p2.u;
        *reinterpret_cast<unsigned int*>(&vlds[(e + 3) * VSTRIDE + 2 * rp]) = p3.u;
      }
    }
    __syncthreads();  // tiles ready

    // ---- Scores: M1 = Q_rows . K_cols^T ; M2 = K_rows . Q_cols^T (= scores^T tile)
    v8f M1[2], M2[2];
    M1[0] = vzero; M1[1] = vzero; M2[0] = vzero; M2[1] = vzero;
#pragma unroll
    for (int na = 0; na < 2; ++na) {
      // B operand (32x16): lane holds col n, k in [hi*8, hi*8+8) and [16+hi*8, 16+hi*8+8) along E
      const _Float16* kb = &klds[(na * 16 + n) * KSTRIDE + hi * 8];
      const _Float16* qb = &qlds[(na * 16 + n) * KSTRIDE + hi * 8];
#pragma unroll
      for (int c = 0; c < 4; ++c) {
        v16h bk = ldsOp(kb + 32 * c, 16);
        M1[na] = __builtin_amdgcn_wmma_f32_16x16x32_f16(
            false, QA[c].v, false, bk, (short)0, M1[na], false, false);
        v16h bq = ldsOp(qb + 32 * c, 16);
        M2[na] = __builtin_amdgcn_wmma_f32_16x16x32_f16(
            false, KA[c].v, false, bq, (short)0, M2[na], false, false);
      }
    }

    // ---- Pointwise: fused sigmoid product; diag cndmask; A_out; entropy ----
#pragma unroll
    for (int na = 0; na < 2; ++na) {
      const int gcol = s0 + na * 16 + n;
#pragma unroll
      for (int j = 0; j < 8; ++j) {
        const int gr = l0 + j + 8 * hi;  // C-layout: lanes 0-15 rows 0-7, 16-31 rows 8-15
        const float sc1 = M1[na][j];
        const float sc2 = M2[na][j];
        // sigm(x)*sigm(y) = 1/((1+e^-x)(1+e^-y)); nit05 = -0.5/tau
        const float e1 = __expf((sc1 + sc2) * nit05);
        const float e2 = __expf((sc1 - sc2) * nit05);
        float att = 1.0f / ((1.0f + e1) * (1.0f + e2));
        if (gr == gcol) att = 0.0f;
        __builtin_nontemporal_store(att, aout_base + (size_t)gr * LQ + gcol);
        ent[j] += att * __logf(fmaxf(att, 1e-8f));
        attlds[wave][(j + 8 * hi) * 32 + na * 16 + n] = (_Float16)att;
      }
    }

    // ---- att (16x32 f16 A-operand from LDS) x V (32x16 f16 B-operands) ----
    // attlds is wave-local: same-wave LDS ops are in-order, no barrier needed.
    v16h attA = ldsOp(&attlds[wave][n * 32 + hi * 8], 16);
#pragma unroll
    for (int t = 0; t < 8; ++t) {
      v16h bv = ldsOp(&vlds[(16 * t + n) * VSTRIDE + hi * 8], 16);
      accV[t] = __builtin_amdgcn_wmma_f32_16x16x32_f16(
          false, attA, false, bv, (short)0, accV[t], false, false);
    }
  }

  // ---- Write V_out (B, L, H, E) ----
#pragma unroll
  for (int t = 0; t < 8; ++t) {
#pragma unroll
    for (int j = 0; j < 8; ++j) {
      const int gr = l0 + j + 8 * hi;
      Vout[(((size_t)b * LQ + gr) * NH + h) * ED + 16 * t + n] = accV[t][j];
    }
  }

  // ---- Entropy: reduce across the 16 lanes of each half, write per row ----
#pragma unroll
  for (int j = 0; j < 8; ++j) {
    float e = ent[j];
    e += __shfl_xor(e, 1);
    e += __shfl_xor(e, 2);
    e += __shfl_xor(e, 4);
    e += __shfl_xor(e, 8);
    ent[j] = -e;
  }
  if (n == 0) {
    float* ep = Ent + (size_t)(b * NH + h) * LQ;
#pragma unroll
    for (int j = 0; j < 8; ++j) ep[l0 + j + 8 * hi] = ent[j];
  }
}

extern "C" void kernel_launch(void* const* d_in, const int* in_sizes, int n_in,
                              void* d_out, int out_size, void* d_ws, size_t ws_size,
                              hipStream_t stream) {
  (void)in_sizes; (void)n_in; (void)out_size; (void)d_ws; (void)ws_size;
  const float* Q  = (const float*)d_in[0];
  const float* K  = (const float*)d_in[1];
  const float* V  = (const float*)d_in[2];
  const float* LT = (const float*)d_in[3];
  float* Vout = (float*)d_out;                            // (B, L, H, E)
  float* Aout = Vout + (size_t)BQ * LQ * NH * ED;         // (B, H, L, L)
  float* Ent  = Aout + (size_t)BQ * NH * LQ * LQ;         // (B, H, L)
  dim3 grid(BQ * NH * (LQ / 64));  // 1024 blocks: one 64-row band of one (b,h)
  toeplitz_attn_kernel<<<grid, 128, 0, stream>>>(Q, K, V, LT, Vout, Aout, Ent);
}
